// SpectralRewiring_57904749084727
// MI455X (gfx1250) — matrix operand order repaired and checked by reading, not compiled, not run on hardware
//
#include <hip/hip_runtime.h>
#include <hip/hip_bf16.h>

// ---- problem constants (match reference) ----
#define NN    16384
#define EE    262144
#define INDIM 512
#define HH    64
#define KK    8
#define NK    (NN * KK)   // 131072

typedef __attribute__((ext_vector_type(16))) __bf16 v16bf;
typedef __attribute__((ext_vector_type(8)))  float  v8f;
typedef int vi4 __attribute__((ext_vector_type(4)));

union BFrag { v16bf v; unsigned u[8]; uint4 q[2]; };

// hardware bf16 convert (RNE)
__device__ __forceinline__ unsigned short f2bf(float f) {
    return __builtin_bit_cast(unsigned short, (__bf16)f);
}
__device__ __forceinline__ unsigned pack2bf(float lo, float hi) {
    return (unsigned)f2bf(lo) | ((unsigned)f2bf(hi) << 16);
}
// exact bf16 -> f32 (single shift, no cvt)
__device__ __forceinline__ float bf2f(unsigned short u) {
    return __builtin_bit_cast(float, (unsigned)u << 16);
}

// ---- CDNA5 async global->LDS copy (ASYNCcnt path), guarded fallback ----
#if defined(__has_builtin)
#  if __has_builtin(__builtin_amdgcn_global_load_async_to_lds_b128) && \
      __has_builtin(__builtin_amdgcn_s_wait_asynccnt)
#    define USE_ASYNC_LDS 1
#  endif
#endif
#ifndef USE_ASYNC_LDS
#  define USE_ASYNC_LDS 0
#endif

__device__ __forceinline__ void async_copy16(const uint4* gsrc, uint4* ldst) {
#if USE_ASYNC_LDS
    __builtin_amdgcn_global_load_async_to_lds_b128(
        (vi4*)gsrc,
        (__attribute__((address_space(3))) vi4*)ldst, 0, 0);
#else
    *ldst = *gsrc;
#endif
}
__device__ __forceinline__ void async_wait0() {
#if USE_ASYNC_LDS
    __builtin_amdgcn_s_wait_asynccnt(0);
#endif
}

// ---------------- utility kernels ----------------
__global__ void fill_f32_kernel(float* __restrict__ p, float v, int n) {
    int i = blockIdx.x * 256 + threadIdx.x;
    if (i < n) p[i] = v;
}

__global__ void cvt_bf16_kernel(const float* __restrict__ src,
                                unsigned short* __restrict__ dst, int n) {
    int i = blockIdx.x * 256 + threadIdx.x;
    if (i < n) dst[i] = f2bf(src[i]);
}

// pre-swizzle a [Kdim x 64] f32 weight matrix into WMMA B-fragment order:
// out dword index = (((ktile*4 + t)*32 + lane)*8 + j), value = packed bf16 pair
// (W[k][t*16+n], W[k+1][t*16+n]) with k = ktile*32 + 2j + 16*(lane>>4), n = lane&15.
__global__ void swz_w_kernel(const float* __restrict__ W, unsigned* __restrict__ out, int Kdim) {
    int i = blockIdx.x * 256 + threadIdx.x;
    int total = (Kdim >> 5) << 10;            // (Kdim/32)*4*32*8
    if (i >= total) return;
    int j     = i & 7;
    int lane  = (i >> 3) & 31;
    int t     = (i >> 8) & 3;
    int ktile = i >> 10;
    int hs = lane >> 4, n = lane & 15;
    int k = ktile * 32 + 2 * j + 16 * hs;
    out[i] = pack2bf(W[k * HH + t * 16 + n], W[(k + 1) * HH + t * 16 + n]);
}

__global__ void degree_kernel(const int* __restrict__ row, float* __restrict__ deg, int e) {
    int i = blockIdx.x * 256 + threadIdx.x;
    if (i < e) atomicAdd(&deg[row[i]], 1.0f);
}

__global__ void dinv_kernel(float* __restrict__ deg, int n) {
    int i = blockIdx.x * 256 + threadIdx.x;
    if (i < n) {
        float d = deg[i];
        deg[i] = (d > 0.0f) ? rsqrtf(d) : 0.0f;   // dinv in place
    }
}

// agg[row] += (dinv[row]*dinv[col]) * g[col]   (edge x feature parallel, g is bf16)
__global__ void msg_kernel(const int* __restrict__ ei, const float* __restrict__ dinv,
                           const unsigned short* __restrict__ gbf, float* __restrict__ agg) {
    int tid = blockIdx.x * 256 + threadIdx.x;
    int e = tid >> 6;
    int h = tid & 63;
    if (e < EE) {
        int r = ei[e];
        int c = ei[EE + e];
        float w = dinv[r] * dinv[c];
        atomicAdd(&agg[r * HH + h], w * bf2f(gbf[c * HH + h]));
    }
}

// filtered = g - agg  -> bf16
__global__ void filtered_kernel(const unsigned short* __restrict__ gbf,
                                const float* __restrict__ agg,
                                unsigned short* __restrict__ fbf, int n) {
    int i = blockIdx.x * 256 + threadIdx.x;
    if (i < n) fbf[i] = f2bf(bf2f(gbf[i]) - agg[i]);
}

// ---------------- WMMA bf16 GEMM:  C[N x 64] = act(A[N x Kdim] @ W[Kdim x 64] + b) ----------------
// A and C in bf16 (row-major), W pre-swizzled into fragment order. All operand
// loads are b128; B fragments for all 4 sub-tiles are loaded before the 4 WMMAs
// so the scheduler can stagger the loadcnt waits.
__global__ __launch_bounds__(128) void gemm_bf16_kernel(
    const unsigned short* __restrict__ Abf, const uint4* __restrict__ Wq,
    const float* __restrict__ bias, unsigned short* __restrict__ Cbf,
    int Kdim, int relu)
{
    const int lane = threadIdx.x & 31;
    const int wave = threadIdx.x >> 5;
    const int rowBase = (blockIdx.x * 4 + wave) * 16;
    const int m  = lane & 15;
    const int hs = lane >> 4;
    const int row = rowBase + m;

    const uint4* Ar = (const uint4*)Abf + row * (Kdim >> 3);   // uint4s per row

    v8f acc[4];
    #pragma unroll
    for (int t = 0; t < 4; ++t)
        #pragma unroll
        for (int r = 0; r < 8; ++r) acc[t][r] = 0.0f;

    for (int ktile = 0; ktile < (Kdim >> 5); ++ktile) {
        BFrag a;
        a.q[0] = Ar[ktile * 4 + 2 * hs];        // dwords ktile*16 + 8hs..
        a.q[1] = Ar[ktile * 4 + 2 * hs + 1];    // dwords ktile*16 + 8hs + 4..
        BFrag b[4];
        #pragma unroll
        for (int t = 0; t < 4; ++t) {
            const uint4* bp = Wq + ((ktile * 4 + t) * 32 + lane) * 2;
            b[t].q[0] = bp[0];
            b[t].q[1] = bp[1];
        }
        #pragma unroll
        for (int t = 0; t < 4; ++t) {
            acc[t] = __builtin_amdgcn_wmma_f32_16x16x32_bf16(
                false, a.v, false, b[t].v, (short)0, acc[t], false, false);
        }
    }

    #pragma unroll
    for (int t = 0; t < 4; ++t) {
        int col = t * 16 + m;
        float bv = bias[col];
        #pragma unroll
        for (int r = 0; r < 8; ++r) {
            float v = acc[t][r] + bv;
            if (relu) v = fmaxf(v, 0.0f);
            Cbf[(rowBase + r + 8 * hs) * HH + col] = f2bf(v);
        }
    }
}

// ---------------- fused sim = F @ F^T  +  running top-8 per row ----------------
// block = 128 threads = 4 waves; block owns 64 rows. Column tiles of F (8 KB) are
// double-buffered in LDS via CDNA5 async global->LDS loads: the fill of tile n+1
// overlaps the WMMA + top-k phase of tile n. Each wave runs 8 WMMAs per tile
// (4 sub-tiles x K=64) and each lane keeps a register-resident sorted top-8 over
// half of each 16-column sub-tile (guarded by a group-max pre-filter).
__global__ __launch_bounds__(128) void simtopk_kernel(
    const unsigned short* __restrict__ Fh, float* __restrict__ out_vals,
    int* __restrict__ out_i)
{
    __shared__ uint4 stage[2][64][9];         // double-buffered, padded vs banks
    __shared__ float ctile[4][4][16][17];     // wave, sub-tile, row, col (padded)
    __shared__ float mv[4][16][16];
    __shared__ int   mi[4][16][16];

    const int tid  = threadIdx.x;
    const int lane = tid & 31;
    const int wave = tid >> 5;
    const int m  = lane & 15;
    const int hs = lane >> 4;
    const int rowBase = blockIdx.x * 64 + wave * 16;

    const uint4* Fq = (const uint4*)Fh;       // 8 x uint4 per 64-feature row

    // A fragments for this wave's 16 rows (K = 0..31, 32..63), loaded once as b128
    BFrag a0, a1;
    {
        const uint4* Fr = Fq + (rowBase + m) * 8;
        a0.q[0] = Fr[0 + 2 * hs];  a0.q[1] = Fr[1 + 2 * hs];
        a1.q[0] = Fr[4 + 2 * hs];  a1.q[1] = Fr[5 + 2 * hs];
    }

    float tv[8]; int ti[8];
    #pragma unroll
    for (int s = 0; s < 8; ++s) { tv[s] = -3.4e38f; ti[s] = 0; }

    const int NT = NN / 64;

    // prologue: stage tile 0 into buffer 0 (async)
    #pragma unroll
    for (int s = 0; s < 4; ++s) {
        int idx = tid + s * 128;              // 0..511
        async_copy16(Fq + (idx >> 3) * 8 + (idx & 7), &stage[0][idx >> 3][idx & 7]);
    }

    for (int nt = 0; nt < NT; ++nt) {
        const int cur = nt & 1;
        const int colBase = nt * 64;

        async_wait0();                         // my wave's fills of stage[cur] done
        __syncthreads();                       // everyone's fills visible

        if (nt + 1 < NT) {                     // kick off tile n+1 into other buffer
            #pragma unroll
            for (int s = 0; s < 4; ++s) {
                int idx = tid + s * 128;
                async_copy16(Fq + (colBase + 64 + (idx >> 3)) * 8 + (idx & 7),
                             &stage[cur ^ 1][idx >> 3][idx & 7]);
            }
        }

        // 4 sub-tiles of 16 columns: 2 WMMAs each (K = 64)
        #pragma unroll
        for (int t = 0; t < 4; ++t) {
            const int col = t * 16 + m;
            BFrag b0, b1;
            b0.q[0] = stage[cur][col][2 * hs];       // dwords 8hs..   (K = 16hs..)
            b0.q[1] = stage[cur][col][2 * hs + 1];
            b1.q[0] = stage[cur][col][4 + 2 * hs];   // +32 K
            b1.q[1] = stage[cur][col][5 + 2 * hs];
            v8f c = {};
            c = __builtin_amdgcn_wmma_f32_16x16x32_bf16(false, a0.v, false, b0.v, (short)0, c, false, false);
            c = __builtin_amdgcn_wmma_f32_16x16x32_bf16(false, a1.v, false, b1.v, (short)0, c, false, false);
            #pragma unroll
            for (int r = 0; r < 8; ++r) ctile[wave][t][r + 8 * hs][m] = c[r];
        }
        __syncthreads();                       // fence ctile writes (and buffer reuse)

        // lane (m, hs) scans columns hs*8..hs*8+7 of row m in each sub-tile
        #pragma unroll
        for (int t = 0; t < 4; ++t) {
            float cv[8];
            #pragma unroll
            for (int jj = 0; jj < 8; ++jj) cv[jj] = ctile[wave][t][m][hs * 8 + jj];
            float cmax = fmaxf(fmaxf(fmaxf(cv[0], cv[1]), fmaxf(cv[2], cv[3])),
                               fmaxf(fmaxf(cv[4], cv[5]), fmaxf(cv[6], cv[7])));
            if (cmax > tv[7]) {                // rare after warmup
                #pragma unroll 1               // keep a single insertion ladder
                for (int jj = 0; jj < 8; ++jj) {
                    float v = ctile[wave][t][m][hs * 8 + jj];
                    if (v > tv[7]) {
                        tv[7] = v; ti[7] = colBase + t * 16 + hs * 8 + jj;
                        #pragma unroll
                        for (int s = 7; s > 0; --s) {   // one bubble pass re-sorts
                            bool sw = tv[s] > tv[s - 1];
                            float va = sw ? tv[s - 1] : tv[s];
                            float vb = sw ? tv[s]     : tv[s - 1];
                            int   ia = sw ? ti[s - 1] : ti[s];
                            int   ib = sw ? ti[s]     : ti[s - 1];
                            tv[s] = va; tv[s - 1] = vb;
                            ti[s] = ia; ti[s - 1] = ib;
                        }
                    }
                }
            }
        }
    }

    // merge the two per-row sorted top-8 lists (2 lanes per row -> 1 lane per row)
    #pragma unroll
    for (int s = 0; s < 8; ++s) { mv[wave][m][hs * 8 + s] = tv[s]; mi[wave][m][hs * 8 + s] = ti[s]; }
    __syncthreads();

    if (lane < 16) {
        int row = rowBase + lane;
        int p = 0, q = 0;
        #pragma unroll
        for (int s = 0; s < 8; ++s) {
            float av = mv[wave][lane][p];
            float bv = mv[wave][lane][8 + q];
            bool ta = av >= bv;
            float v = ta ? av : bv;
            int  id = ta ? mi[wave][lane][p] : mi[wave][lane][8 + q];
            p += ta ? 1 : 0;
            q += ta ? 0 : 1;
            out_vals[row * KK + s]       = v;      // topk_vals (float32)
            out_i[NK + row * KK + s]     = row;    // new_edge_index[0] = src
            out_i[2 * NK + row * KK + s] = id;     // new_edge_index[1] = topk idx
        }
    }
}

// ---------------- launch ----------------
extern "C" void kernel_launch(void* const* d_in, const int* in_sizes, int n_in,
                              void* d_out, int out_size, void* d_ws, size_t ws_size,
                              hipStream_t stream) {
    const float* x   = (const float*)d_in[0];
    const int*   ei  = (const int*)  d_in[1];
    const float* W1  = (const float*)d_in[2];  const float* b1  = (const float*)d_in[3];
    const float* W2  = (const float*)d_in[4];  const float* b2  = (const float*)d_in[5];
    const float* W3  = (const float*)d_in[6];  const float* b3  = (const float*)d_in[7];
    const float* W4  = (const float*)d_in[8];  const float* b4  = (const float*)d_in[9];
    const float* Wg1 = (const float*)d_in[10]; const float* bg1 = (const float*)d_in[11];
    const float* Wg2 = (const float*)d_in[12]; const float* bg2 = (const float*)d_in[13];

    char* ws = (char*)d_ws;
    float*          deg = (float*)(ws);                         // N f32 (becomes dinv)
    unsigned short* xbf = (unsigned short*)(ws + 65536);        // N*512 bf16 (16 MB)
    unsigned short* hb0 = (unsigned short*)(ws + 16842752);     // N*64 bf16 (2 MB)
    unsigned short* hb1 = (unsigned short*)(ws + 18939904);     // N*64 bf16 (2 MB)
    float*          agg = (float*)(ws + 21037056);              // N*64 f32 (4 MB)
    unsigned short* fbf = (unsigned short*)(ws + 25231360);     // N*64 bf16 (2 MB)
    unsigned*       Wsw = (unsigned*)(ws + 27328512);           // swizzled weights

    // swizzled-weight sub-offsets (dwords): W1=16384, then 5 x 2048
    unsigned* Ws1  = Wsw;
    unsigned* Ws2  = Wsw + 16384;
    unsigned* Ws3  = Wsw + 18432;
    unsigned* Ws4  = Wsw + 20480;
    unsigned* Wsg1 = Wsw + 22528;
    unsigned* Wsg2 = Wsw + 24576;

    // zero accumulators (ws is poisoned; atomics accumulate)
    fill_f32_kernel<<<(NN + 255) / 256, 256, 0, stream>>>(deg, 0.0f, NN);
    fill_f32_kernel<<<(NN * HH + 255) / 256, 256, 0, stream>>>(agg, 0.0f, NN * HH);

    // x -> bf16 ; weights -> swizzled bf16 fragments
    cvt_bf16_kernel<<<(NN * INDIM + 255) / 256, 256, 0, stream>>>(x, xbf, NN * INDIM);
    swz_w_kernel<<<(16384 + 255) / 256, 256, 0, stream>>>(W1,  Ws1,  INDIM);
    swz_w_kernel<<<(2048  + 255) / 256, 256, 0, stream>>>(W2,  Ws2,  HH);
    swz_w_kernel<<<(2048  + 255) / 256, 256, 0, stream>>>(W3,  Ws3,  HH);
    swz_w_kernel<<<(2048  + 255) / 256, 256, 0, stream>>>(W4,  Ws4,  HH);
    swz_w_kernel<<<(2048  + 255) / 256, 256, 0, stream>>>(Wg1, Wsg1, HH);
    swz_w_kernel<<<(2048  + 255) / 256, 256, 0, stream>>>(Wg2, Wsg2, HH);

    // degrees -> dinv
    degree_kernel<<<(EE + 255) / 256, 256, 0, stream>>>(ei, deg, EE);
    dinv_kernel<<<(NN + 255) / 256, 256, 0, stream>>>(deg, NN);

    // MLP stack, all-bf16 activations (g ends in hb1)
    gemm_bf16_kernel<<<256, 128, 0, stream>>>(xbf, (const uint4*)Ws1,  b1,  hb0, INDIM, 1);
    gemm_bf16_kernel<<<256, 128, 0, stream>>>(hb0, (const uint4*)Ws2,  b2,  hb1, HH,    0);
    gemm_bf16_kernel<<<256, 128, 0, stream>>>(hb1, (const uint4*)Ws3,  b3,  hb0, HH,    1);
    gemm_bf16_kernel<<<256, 128, 0, stream>>>(hb0, (const uint4*)Ws4,  b4,  hb1, HH,    0);
    gemm_bf16_kernel<<<256, 128, 0, stream>>>(hb1, (const uint4*)Wsg1, bg1, hb0, HH,    1);
    gemm_bf16_kernel<<<256, 128, 0, stream>>>(hb0, (const uint4*)Wsg2, bg2, hb1, HH,    0);

    // graph aggregation, filtered = g - agg (bf16)
    msg_kernel<<<(EE * HH) / 256, 256, 0, stream>>>(ei, deg, hb1, agg);
    filtered_kernel<<<(NN * HH + 255) / 256, 256, 0, stream>>>(hb1, agg, fbf, NN * HH);

    // fused sim + top-8
    simtopk_kernel<<<NN / 64, 128, 0, stream>>>(fbf, (float*)d_out, (int*)d_out);
}
